// YoloNASPostprocessingModule_79388175499539
// MI455X (gfx1250) — matrix operand order repaired and checked by reading, not compiled, not run on hardware
//
#include <hip/hip_runtime.h>
#include <stdint.h>

#define N_ANCHORS 262144
#define N_CLASSES 80
#define K_TOP     1024
#define SCORE_TH  0.5f
#define NMS_TH    0.6f
#define A_ROWS    128
#define LIST_CAP  4096

typedef unsigned int v4u  __attribute__((ext_vector_type(4)));
typedef int          v8i  __attribute__((ext_vector_type(8)));
typedef int          v4i  __attribute__((ext_vector_type(4)));
typedef _Float16     v16h __attribute__((ext_vector_type(16)));
typedef float        v8f  __attribute__((ext_vector_type(8)));

// ---------------------------------------------------------------------------
// Stage A: per-anchor max/argmax over 80 classes, fused with the confidence
// histogram. Score tile (128x80 f32 = 40KB) staged into LDS via the Tensor
// Data Mover (async, TENSORcnt), then each thread reduces its row from LDS
// with float4 (ds_load_b128) accesses.
// ---------------------------------------------------------------------------
__global__ __launch_bounds__(A_ROWS) void argmax_hist_kernel(
    const float* __restrict__ scores, float* __restrict__ conf,
    int* __restrict__ label, int* __restrict__ bins) {
  __shared__ float tile[A_ROWS * N_CLASSES];
  const int block = blockIdx.x;
#if __has_builtin(__builtin_amdgcn_tensor_load_to_lds)
  if (threadIdx.x < 32) {  // one wave issues the TDM descriptor
    uint64_t gaddr =
        (uint64_t)(uintptr_t)(scores + (size_t)block * A_ROWS * N_CLASSES);
    v4u g0;
    g0.x = 1u;  // count=1 (valid user descriptor)
    g0.y = 0u;  // lds_addr = 0 (tile is the only LDS object)
    g0.z = (unsigned)(gaddr & 0xFFFFFFFFu);
    g0.w = (unsigned)((gaddr >> 32) & 0x01FFFFFFu) | (2u << 30);  // type=2
    v8i g1;
    g1[0] = (int)(2u << 16);          // data_size = 4B
    g1[1] = (int)(N_CLASSES << 16);   // tensor_dim0 lo16
    g1[2] = (int)(A_ROWS << 16);      // tensor_dim0 hi16=0 | tensor_dim1 lo16
    g1[3] = (int)(N_CLASSES << 16);   // tensor_dim1 hi16=0 | tile_dim0
    g1[4] = A_ROWS;                   // tile_dim1 | tile_dim2=0
    g1[5] = N_CLASSES;                // tensor_dim0_stride lo32
    g1[6] = 0;
    g1[7] = 0;
    v4i g2 = {0, 0, 0, 0};
    v4i g3 = {0, 0, 0, 0};
#if __clang_major__ >= 23
    v8i gx = {0, 0, 0, 0, 0, 0, 0, 0};
    __builtin_amdgcn_tensor_load_to_lds(g0, g1, g2, g3, gx, 0);
#else
    __builtin_amdgcn_tensor_load_to_lds(g0, g1, g2, g3, 0);
#endif
    __builtin_amdgcn_s_wait_tensorcnt(0);
  }
  __syncthreads();
#else
  for (int i = threadIdx.x; i < A_ROWS * N_CLASSES; i += blockDim.x)
    tile[i] = scores[(size_t)block * A_ROWS * N_CLASSES + i];
  __syncthreads();
#endif
  const int t = threadIdx.x;
  const float4* row = (const float4*)(tile + t * N_CLASSES);  // 320B aligned
  float best = -1e30f;
  int bi = 0;
#pragma unroll
  for (int g = 0; g < N_CLASSES / 4; ++g) {
    float4 v = row[g];
    // strict > keeps first occurrence (argmax semantics)
    if (v.x > best) { best = v.x; bi = 4 * g + 0; }
    if (v.y > best) { best = v.y; bi = 4 * g + 1; }
    if (v.z > best) { best = v.z; bi = 4 * g + 2; }
    if (v.w > best) { best = v.w; bi = 4 * g + 3; }
  }
  const int a = block * A_ROWS + t;
  const bool pass = (best >= SCORE_TH);
  conf[a] = pass ? best : -1.0f;
  label[a] = bi;
  if (pass) {
    unsigned bits = __float_as_uint(best);
    unsigned bin = (bits - 0x3F000000u) >> 11;  // monotonic on [0.5,1)
    if (bin > 4095u) bin = 4095u;
    atomicAdd(&bins[bin], 1);
  }
}

// ---------------------------------------------------------------------------
// Top-K selection: threshold bin from histogram -> compaction -> exact
// bitonic sort of 1024 records.
// ---------------------------------------------------------------------------
__global__ void zero_kernel(int* bins, int* meta) {
  int t = blockIdx.x * blockDim.x + threadIdx.x;
  if (t < 4096) bins[t] = 0;
  if (t < 8) meta[t] = 0;
}

// meta: [0]=T threshold bin, [1]=need from bin T, [2]=topBase, [3]=ctrA, [4]=ctrT
__global__ void scan_kernel(const int* __restrict__ bins, int* __restrict__ meta) {
  if (threadIdx.x != 0 || blockIdx.x != 0) return;
  int run = 0, T = -1, need = 0, topBase = 0;
  for (int b = 4095; b >= 0; --b) {
    int base = run;
    run += bins[b];
    if (T < 0 && run >= K_TOP) { T = b; need = K_TOP - base; topBase = base; }
  }
  if (T < 0) topBase = run;  // fewer than K_TOP candidates total
  meta[0] = T; meta[1] = need; meta[2] = topBase;
}

__global__ void compact_kernel(const float* __restrict__ conf,
                               const int* __restrict__ label,
                               int* __restrict__ meta,
                               float* __restrict__ selC, int* __restrict__ selL,
                               int* __restrict__ selI,
                               float* __restrict__ listC, int* __restrict__ listI) {
  int a = blockIdx.x * blockDim.x + threadIdx.x;
  if (a >= N_ANCHORS) return;
  float c = conf[a];
  if (c < SCORE_TH) return;
  unsigned bits = __float_as_uint(c);
  int bin = (int)((bits - 0x3F000000u) >> 11);
  if (bin > 4095) bin = 4095;
  int T = meta[0];
  if (bin > T) {
    int s = atomicAdd(&meta[3], 1);  // slot set is deterministic; order fixed by sort
    selC[s] = c; selL[s] = label[a]; selI[s] = a;
  } else if (bin == T) {
    int s = atomicAdd(&meta[4], 1);
    if (s < LIST_CAP) { listC[s] = c; listI[s] = a; }
  }
}

__global__ __launch_bounds__(K_TOP) void select_sort_kernel(
    const int* __restrict__ meta, const float* __restrict__ listC,
    const int* __restrict__ listI, const int* __restrict__ label,
    float* __restrict__ selC, int* __restrict__ selL, int* __restrict__ selI,
    const float* __restrict__ bboxes, float* __restrict__ topC,
    int* __restrict__ topL, int* __restrict__ topI, float* __restrict__ topB) {
  __shared__ float cf[K_TOP];
  __shared__ int id[K_TOP];
  __shared__ int lb[K_TOP];
  const int t = threadIdx.x;
  int need = meta[1], topBase = meta[2];
  int cntT = meta[4];
  if (cntT > LIST_CAP) cntT = LIST_CAP;
  if (need > cntT) need = cntT;
  if (t >= topBase) { selC[t] = -1.0f; selL[t] = 0; selI[t] = 0; }
  __syncthreads();
  // rank-select the boundary-bin entries by (conf desc, index asc)
  for (int e = t; e < cntT; e += K_TOP) {
    float mc = listC[e]; int mi = listI[e];
    int rank = 0;
    for (int q = 0; q < cntT; ++q) {
      float qc = listC[q]; int qi = listI[q];
      rank += (qc > mc) || (qc == mc && qi < mi);
    }
    if (rank < need) {
      int s = topBase + rank;
      selC[s] = mc; selL[s] = label[mi]; selI[s] = mi;
    }
  }
  __syncthreads();
  cf[t] = selC[t]; id[t] = selI[t]; lb[t] = selL[t];
  __syncthreads();
  // bitonic sort: conf descending, index ascending on ties
  for (int size = 2; size <= K_TOP; size <<= 1) {
    for (int stride = size >> 1; stride > 0; stride >>= 1) {
      int p = t ^ stride;
      if (p > t) {
        bool up = (t & size) != 0;
        bool tFirst = (cf[t] > cf[p]) || (cf[t] == cf[p] && id[t] < id[p]);
        bool doSwap = up ? tFirst : !tFirst;
        if (doSwap) {
          float c0 = cf[t]; cf[t] = cf[p]; cf[p] = c0;
          int i0 = id[t]; id[t] = id[p]; id[p] = i0;
          int l0 = lb[t]; lb[t] = lb[p]; lb[p] = l0;
        }
      }
      __syncthreads();
    }
  }
  topC[t] = cf[t]; topL[t] = lb[t]; topI[t] = id[t];
  ((float4*)topB)[t] = ((const float4*)bboxes)[id[t]];
}

// ---------------------------------------------------------------------------
// Stage C: suppression matrix. Same-class mask = onehot(label) @ onehot^T via
// v_wmma_f32_16x16x32_f16 (K padded 80->96, 3 chunks), fused with pairwise
// IoU. One wave per 16x16 tile of the 1024x1024 matrix.
// Fragment layouts per the ISA (§7.12.2 / §7.12.4):
//   A (16x32): lane<16 -> M=lane,    K[h] = h + (h>=8 ? 8 : 0)
//              lane>=16 -> M=lane-16, K[h] = h + 8 + (h>=8 ? 8 : 0)
//   B (32x16): lane<16 -> N=lane,    K[h] = h + (h>=8 ? 8 : 0)
//              lane>=16 -> N=lane-16, K[h] = h + 8 + (h>=8 ? 8 : 0)
// ---------------------------------------------------------------------------
__global__ __launch_bounds__(32) void sameclass_iou_kernel(
    const int* __restrict__ topL, const float* __restrict__ topB,
    unsigned char* __restrict__ sup) {
  const int lane = threadIdx.x;
  const int ti = blockIdx.x, tj = blockIdx.y;
  const int hs = (lane >= 16) ? 1 : 0;
  const int la = topL[ti * 16 + (lane & 15)];
  const int lb = topL[tj * 16 + (lane & 15)];
  v8f c = {0.f, 0.f, 0.f, 0.f, 0.f, 0.f, 0.f, 0.f};
#pragma unroll
  for (int chunk = 0; chunk < 3; ++chunk) {
    const int kbase = chunk * 32;
    v16h a, b;
#pragma unroll
    for (int h = 0; h < 16; ++h) {
      int kgA = kbase + h + ((h >= 8) ? 8 : 0) + (hs ? 8 : 0);
      int kgB = kbase + h + ((h >= 8) ? 8 : 0) + (hs ? 8 : 0);
      a[h] = (kgA == la) ? (_Float16)1 : (_Float16)0;
      b[h] = (kgB == lb) ? (_Float16)1 : (_Float16)0;
    }
    c = __builtin_amdgcn_wmma_f32_16x16x32_f16(false, a, false, b, (short)0, c,
                                               false, false);
  }
  // D layout: VGPR r -> M = r + 8*hs, N = lane&15
  const int j = tj * 16 + (lane & 15);
  const float4 bj = ((const float4*)topB)[j];
  const float areaJ = (bj.z - bj.x) * (bj.w - bj.y);
#pragma unroll
  for (int r = 0; r < 8; ++r) {
    const int i = ti * 16 + r + hs * 8;
    const float4 bi = ((const float4*)topB)[i];
    const float areaI = (bi.z - bi.x) * (bi.w - bi.y);
    float iw = fminf(bi.z, bj.z) - fmaxf(bi.x, bj.x);
    float ih = fminf(bi.w, bj.w) - fmaxf(bi.y, bj.y);
    iw = fmaxf(iw, 0.0f); ih = fmaxf(ih, 0.0f);
    const float inter = iw * ih;
    const float uni = areaI + areaJ - inter;
    const float iou = inter / fmaxf(uni, 1e-9f);
    const bool same = c[r] > 0.5f;
    sup[(size_t)i * K_TOP + j] = (same && iou > NMS_TH) ? 1 : 0;
  }
}

// ---------------------------------------------------------------------------
// Stage D: sequential greedy NMS over the precomputed matrix + masked output.
// ---------------------------------------------------------------------------
__global__ __launch_bounds__(K_TOP) void nms_out_kernel(
    const unsigned char* __restrict__ sup, const float* __restrict__ topC,
    const int* __restrict__ topL, const float* __restrict__ topB,
    float* __restrict__ out) {
  __shared__ int keep[K_TOP];
  const int t = threadIdx.x;
  keep[t] = (topC[t] >= SCORE_TH) ? 1 : 0;
  __syncthreads();
  for (int i = 0; i < K_TOP; ++i) {
    if (keep[i]) {
      if (t > i && sup[(size_t)i * K_TOP + t]) keep[t] = 0;
    }
    __syncthreads();
  }
  const float4 b = ((const float4*)topB)[t];
  const float k = keep[t] ? 1.0f : 0.0f;
  out[t * 6 + 0] = b.x * k;
  out[t * 6 + 1] = b.y * k;
  out[t * 6 + 2] = b.z * k;
  out[t * 6 + 3] = b.w * k;
  out[t * 6 + 4] = topC[t] * k;
  out[t * 6 + 5] = (float)topL[t] * k;
}

// ---------------------------------------------------------------------------
extern "C" void kernel_launch(void* const* d_in, const int* in_sizes, int n_in,
                              void* d_out, int out_size, void* d_ws,
                              size_t ws_size, hipStream_t stream) {
  (void)in_sizes; (void)n_in; (void)out_size; (void)ws_size;
  const float* bboxes = (const float*)d_in[0];  // [N,4]
  const float* scores = (const float*)d_in[1];  // [N,80]
  float* out = (float*)d_out;                   // [1024,6]

  char* w = (char*)d_ws;
  const size_t OFF_CONF  = 0;
  const size_t OFF_LABEL = OFF_CONF  + (size_t)N_ANCHORS * 4;
  const size_t OFF_BINS  = OFF_LABEL + (size_t)N_ANCHORS * 4;
  const size_t OFF_META  = OFF_BINS  + 4096 * 4;
  const size_t OFF_SELC  = OFF_META  + 64;
  const size_t OFF_SELL  = OFF_SELC  + K_TOP * 4;
  const size_t OFF_SELI  = OFF_SELL  + K_TOP * 4;
  const size_t OFF_LISTC = OFF_SELI  + K_TOP * 4;
  const size_t OFF_LISTI = OFF_LISTC + LIST_CAP * 4;
  const size_t OFF_TOPC  = OFF_LISTI + LIST_CAP * 4;
  const size_t OFF_TOPL  = OFF_TOPC  + K_TOP * 4;
  const size_t OFF_TOPI  = OFF_TOPL  + K_TOP * 4;
  const size_t OFF_TOPB  = OFF_TOPI  + K_TOP * 4;
  const size_t OFF_SUP   = OFF_TOPB  + (size_t)K_TOP * 16;

  float* conf  = (float*)(w + OFF_CONF);
  int*   label = (int*)(w + OFF_LABEL);
  int*   bins  = (int*)(w + OFF_BINS);
  int*   meta  = (int*)(w + OFF_META);
  float* selC  = (float*)(w + OFF_SELC);
  int*   selL  = (int*)(w + OFF_SELL);
  int*   selI  = (int*)(w + OFF_SELI);
  float* listC = (float*)(w + OFF_LISTC);
  int*   listI = (int*)(w + OFF_LISTI);
  float* topC  = (float*)(w + OFF_TOPC);
  int*   topL  = (int*)(w + OFF_TOPL);
  int*   topI  = (int*)(w + OFF_TOPI);
  float* topB  = (float*)(w + OFF_TOPB);
  unsigned char* sup = (unsigned char*)(w + OFF_SUP);

  zero_kernel<<<16, 256, 0, stream>>>(bins, meta);
  argmax_hist_kernel<<<N_ANCHORS / A_ROWS, A_ROWS, 0, stream>>>(scores, conf,
                                                                label, bins);
  scan_kernel<<<1, 1, 0, stream>>>(bins, meta);
  compact_kernel<<<N_ANCHORS / 256, 256, 0, stream>>>(conf, label, meta, selC,
                                                      selL, selI, listC, listI);
  select_sort_kernel<<<1, K_TOP, 0, stream>>>(meta, listC, listI, label, selC,
                                              selL, selI, bboxes, topC, topL,
                                              topI, topB);
  sameclass_iou_kernel<<<dim3(K_TOP / 16, K_TOP / 16), 32, 0, stream>>>(topL,
                                                                        topB,
                                                                        sup);
  nms_out_kernel<<<1, K_TOP, 0, stream>>>(sup, topC, topL, topB, out);
}
